// MultiHeadAttention_73126113182062
// MI455X (gfx1250) — matrix operand order repaired and checked by reading, not compiled, not run on hardware
//
#include <hip/hip_runtime.h>

typedef _Float16 v8h  __attribute__((ext_vector_type(8)));
typedef _Float16 v16h __attribute__((ext_vector_type(16)));
typedef float    v8f  __attribute__((ext_vector_type(8)));
typedef unsigned int u32x4 __attribute__((ext_vector_type(4)));
typedef int          i32x4 __attribute__((ext_vector_type(4)));
typedef int          i32x8 __attribute__((ext_vector_type(8)));

#if __has_builtin(__builtin_amdgcn_tensor_load_to_lds) && __has_builtin(__builtin_amdgcn_s_wait_tensorcnt)
#define USE_TDM 1
#else
#define USE_TDM 0
#endif

__device__ __forceinline__ v8f wmma_f16(v16h a, v16h b, v8f c) {
  // (neg_a, A, neg_b, B, c_mod, C, reuse_a, reuse_b)
  return __builtin_amdgcn_wmma_f32_16x16x32_f16(false, a, false, b, (short)0, c, false, false);
}

// A-fragment (16x32 f16): lane = row m (l&15), g = l>>4.
// elements e<8 -> K = g*8+e ; e>=8 -> K = g*8+16+(e-8). Two aligned 16B chunks.
__device__ __forceinline__ v16h load_a_frag(const _Float16* row, int g) {
  v8h lo = *(const v8h*)(row + g * 8);
  v8h hi = *(const v8h*)(row + g * 8 + 16);
  v16h r;
#pragma unroll
  for (int i = 0; i < 8; ++i) { r[i] = lo[i]; r[i + 8] = hi[i]; }
  return r;
}

// B-fragment (32x16 f16): lane = column n (l&15), g = l>>4.
// elements e<8 -> K = g*8+e ; e>=8 -> K = g*8+16+(e-8). Two aligned 16B chunks.
__device__ __forceinline__ v16h load_b_frag(const _Float16* row, int g) {
  v8h lo = *(const v8h*)(row + g * 8);
  v8h hi = *(const v8h*)(row + g * 8 + 16);
  v16h r;
#pragma unroll
  for (int i = 0; i < 8; ++i) { r[i] = lo[i]; r[i + 8] = hi[i]; }
  return r;
}

// ---------------------------------------------------------------------------
// Tiled WMMA GEMM: C[MxN] = A[MxK] * B[KxN].  Block tile 128x128x32,
// 128 threads = 4 waves, each wave computes 64x64 (4x4 WMMA tiles)
// -> 16 WMMAs per 8 LDS fragment loads per K-step.
// A staged fp32/f16 -> f16 row-major; B staged transposed Bs[n][k].
// ---------------------------------------------------------------------------
template <typename TA, typename TC>
__global__ __launch_bounds__(128)
void gemm_wmma_kernel(const TA* __restrict__ A, const float* __restrict__ B,
                      TC* __restrict__ C, int M, int N, int K) {
  constexpr int BK = 32, LDA = 40, LDB = 40;  // row stride 80B (16B-aligned)
  __shared__ _Float16 As[128 * LDA];
  __shared__ _Float16 Bs[128 * LDB];
  const int tid  = threadIdx.x;
  const int lane = tid & 31, wv = tid >> 5;     // 4 waves
  const int m16  = lane & 15, g = lane >> 4;
  const int wm   = (wv & 1) * 64, wn = (wv >> 1) * 64;
  const int m0   = blockIdx.y * 128, n0 = blockIdx.x * 128;

  v8f acc[4][4] = {};

  for (int k0 = 0; k0 < K; k0 += BK) {
    // prefetch next K-step tiles into L2 (global_prefetch_b8)
    if (k0 + BK < K) {
      __builtin_prefetch(A + (size_t)(m0 + tid) * K + (k0 + BK), 0, 0);
      if (tid < BK) __builtin_prefetch(B + (size_t)(k0 + BK + tid) * N + n0, 0, 0);
    }
    // stage A: 128x32, 4 elems/thread/iter (coalesced 16B fp32 reads)
#pragma unroll
    for (int i = 0; i < 8; ++i) {
      int idx = (tid + i * 128) * 4;
      int r = idx >> 5, c = idx & 31;
      const TA* src = A + (size_t)(m0 + r) * K + k0 + c;
      _Float16* dst = &As[r * LDA + c];
      dst[0] = (_Float16)src[0]; dst[1] = (_Float16)src[1];
      dst[2] = (_Float16)src[2]; dst[3] = (_Float16)src[3];
    }
    // stage B: 32x128 fp32, transposed into Bs[n][k]
#pragma unroll
    for (int i = 0; i < 8; ++i) {
      int idx = (tid + i * 128) * 4;
      int kk = idx >> 7, n = idx & 127;
      const float* src = B + (size_t)(k0 + kk) * N + n0 + n;
      Bs[(n + 0) * LDB + kk] = (_Float16)src[0];
      Bs[(n + 1) * LDB + kk] = (_Float16)src[1];
      Bs[(n + 2) * LDB + kk] = (_Float16)src[2];
      Bs[(n + 3) * LDB + kk] = (_Float16)src[3];
    }
    __syncthreads();

    v16h af[4], bf[4];
#pragma unroll
    for (int mi = 0; mi < 4; ++mi)
      af[mi] = load_a_frag(&As[(wm + mi * 16 + m16) * LDA], g);
#pragma unroll
    for (int ni = 0; ni < 4; ++ni)
      bf[ni] = load_b_frag(&Bs[(wn + ni * 16 + m16) * LDB], g);
#pragma unroll
    for (int mi = 0; mi < 4; ++mi)
#pragma unroll
      for (int ni = 0; ni < 4; ++ni)
        acc[mi][ni] = wmma_f16(af[mi], bf[ni], acc[mi][ni]);
    __syncthreads();
  }

  // epilogue: C layout = lane holds column n=l&15; VGPR r holds row r+8*g
#pragma unroll
  for (int mi = 0; mi < 4; ++mi)
#pragma unroll
    for (int ni = 0; ni < 4; ++ni) {
      int ncol = n0 + wn + ni * 16 + m16;
#pragma unroll
      for (int r = 0; r < 8; ++r) {
        int mrow = m0 + wm + mi * 16 + r + 8 * g;
        C[(size_t)mrow * N + ncol] = (TC)acc[mi][ni][r];
      }
    }
}

// ---------------------------------------------------------------------------
// Flash attention, causal.  Grid (T/64, B*H), 128 threads = 4 waves.
// K tile staged by the Tensor Data Mover (strided 64x64 f16 tile with LDS
// row padding), V staged transposed by VALU.  qkv: [B*T][3*1024].
// ---------------------------------------------------------------------------
__global__ __launch_bounds__(128)
void flash_attn_kernel(const _Float16* __restrict__ qkv, _Float16* __restrict__ o) {
  constexpr int T = 2048, Dh = 64, C3 = 3072, Dm = 1024;
  constexpr int LDK = 72;  // row stride 144B = 128B tile row + 16B pad
  __shared__ _Float16 Ks[64 * LDK];
  __shared__ _Float16 Vs[64 * LDK];
  __shared__ _Float16 Ps[4 * 16 * LDK];
  const int tid  = threadIdx.x;
  const int lane = tid & 31, w = tid >> 5;
  const int m16  = lane & 15, g = lane >> 4;
  const int bx   = blockIdx.x;
  const int b    = blockIdx.y >> 4, h = blockIdx.y & 15;
  const float scale = 0.125f;  // 1/sqrt(64)

  // persistent Q A-fragments (Dh=64 -> two 32-wide contraction chunks)
  const int qr = bx * 64 + w * 16 + m16;
  const _Float16* qrow = qkv + (size_t)(b * T + qr) * C3 + h * Dh;
  v16h qf[2];
#pragma unroll
  for (int c = 0; c < 2; ++c) qf[c] = load_a_frag(qrow + c * 32, g);

  float m_run[8], l_run[8];
#pragma unroll
  for (int r = 0; r < 8; ++r) { m_run[r] = -1e30f; l_run[r] = 0.f; }
  v8f oacc[4] = {};

  for (int kt = 0; kt <= bx; ++kt) {
    __syncthreads();
    const _Float16* ksrc = qkv + (size_t)(b * T + kt * 64) * C3 + Dm + h * Dh;
#if USE_TDM
    if (w == 0) {
      // Tensor DMA: 64x64 f16 tile, global row stride 3072 elems, pad LDS
      // rows by 4 DWORDs every 32 DWORDs (-> LDK=72 stride).
      unsigned long long ga = (unsigned long long)ksrc;
      unsigned lds = (unsigned)(unsigned long long)&Ks[0];
      u32x4 g0;
      g0[0] = 1u;                                   // count=1, user mode
      g0[1] = lds;                                  // lds_addr
      g0[2] = (unsigned)ga;                         // global_addr[31:0]
      g0[3] = (unsigned)((ga >> 32) & 0x1FFFFFFu) | (2u << 30);  // [56:32] | type=2
      i32x8 g1;
      g1[0] = (1 << 16) | (1 << 20) | (4 << 22) | (3 << 25); // 2B elems, pad on,
                                                             // interval=32dw, amount=4dw
      g1[1] = 64 << 16;     // tensor_dim0 = 64 (bits 79:48, low half)
      g1[2] = 64 << 16;     // tensor_dim0 hi=0 | tensor_dim1 lo = 64
      g1[3] = 64 << 16;     // tensor_dim1 hi=0 | tile_dim0 = 64
      g1[4] = 64;           // tile_dim1 = 64, tile_dim2 = 0
      g1[5] = 3072;         // tensor_dim0_stride[31:0]
      g1[6] = 0;            // stride hi | dim1_stride lo
      g1[7] = 0;
      i32x4 z4 = {0, 0, 0, 0};
#if defined(__clang_major__) && (__clang_major__ >= 23)
      i32x8 z8 = {0, 0, 0, 0, 0, 0, 0, 0};
      __builtin_amdgcn_tensor_load_to_lds(g0, g1, z4, z4, z8, 0);
#else
      __builtin_amdgcn_tensor_load_to_lds(g0, g1, z4, z4, 0);
#endif
      __builtin_amdgcn_s_wait_tensorcnt(0);
    }
#else
#pragma unroll
    for (int i = 0; i < 4; ++i) {
      int idx = (tid + i * 128) * 8;
      int key = idx >> 6, d = idx & 63;
      *(v8h*)&Ks[key * LDK + d] = *(const v8h*)(ksrc + (size_t)key * C3 + d);
    }
#endif
    // stage V tile transposed: Vs[d][key]
#pragma unroll
    for (int i = 0; i < 4; ++i) {
      int idx = (tid + i * 128) * 8;
      int key = idx >> 6, d = idx & 63;
      v8h val = *(const v8h*)(qkv + (size_t)(b * T + kt * 64 + key) * C3 + 2 * Dm + h * Dh + d);
#pragma unroll
      for (int j = 0; j < 8; ++j) Vs[(d + j) * LDK + key] = val[j];
    }
    __syncthreads();

    // S = Q * K^T : B-fragment column = key, contraction = d (rows of Ks)
    v8f sacc[4];
#pragma unroll
    for (int nt = 0; nt < 4; ++nt) {
      v8f s = {};
#pragma unroll
      for (int c = 0; c < 2; ++c)
        s = wmma_f16(qf[c], load_b_frag(&Ks[(nt * 16 + m16) * LDK + c * 32], g), s);
      sacc[nt] = s;
    }

    // causal mask (diagonal tile only) + scale + row max
    const bool diag = (kt == bx);
    float rowmax[8];
#pragma unroll
    for (int r = 0; r < 8; ++r) rowmax[r] = -1e30f;
#pragma unroll
    for (int nt = 0; nt < 4; ++nt)
#pragma unroll
      for (int r = 0; r < 8; ++r) {
        float v = sacc[nt][r] * scale;
        if (diag) {
          int kg = kt * 64 + nt * 16 + m16;
          int qg = bx * 64 + w * 16 + r + 8 * g;
          if (kg > qg) v = -1e30f;
        }
        sacc[nt][r] = v;
        rowmax[r] = fmaxf(rowmax[r], v);
      }
#pragma unroll
    for (int r = 0; r < 8; ++r)
#pragma unroll
      for (int off = 1; off < 16; off <<= 1)
        rowmax[r] = fmaxf(rowmax[r], __shfl_xor(rowmax[r], off, 32));

    float alpha[8];
#pragma unroll
    for (int r = 0; r < 8; ++r) {
      float mn = fmaxf(m_run[r], rowmax[r]);
      alpha[r] = __expf(m_run[r] - mn);
      m_run[r] = mn;
      l_run[r] *= alpha[r];
    }
    float rowsum[8] = {0.f, 0.f, 0.f, 0.f, 0.f, 0.f, 0.f, 0.f};
#pragma unroll
    for (int nt = 0; nt < 4; ++nt)
#pragma unroll
      for (int r = 0; r < 8; ++r) {
        float p = __expf(sacc[nt][r] - m_run[r]);
        sacc[nt][r] = p;
        rowsum[r] += p;
      }
#pragma unroll
    for (int r = 0; r < 8; ++r) {
#pragma unroll
      for (int off = 1; off < 16; off <<= 1)
        rowsum[r] += __shfl_xor(rowsum[r], off, 32);
      l_run[r] += rowsum[r];
    }
#pragma unroll
    for (int nt = 0; nt < 4; ++nt)
#pragma unroll
      for (int r = 0; r < 8; ++r) oacc[nt][r] *= alpha[r];

    // relayout P: C-layout regs -> per-wave LDS slice -> A-fragments
#pragma unroll
    for (int nt = 0; nt < 4; ++nt)
#pragma unroll
      for (int r = 0; r < 8; ++r)
        Ps[(w * 16 + r + 8 * g) * LDK + nt * 16 + m16] = (_Float16)sacc[nt][r];

    v16h pf[2];
#pragma unroll
    for (int c = 0; c < 2; ++c)
      pf[c] = load_a_frag(&Ps[(w * 16 + m16) * LDK + c * 32], g);

    // O += P * V : B-fragment column = d, contraction = key (rows of Vs)
#pragma unroll
    for (int nt = 0; nt < 4; ++nt)
#pragma unroll
      for (int c = 0; c < 2; ++c)
        oacc[nt] = wmma_f16(pf[c], load_b_frag(&Vs[(nt * 16 + m16) * LDK + c * 32], g), oacc[nt]);
  }

  // normalize and write attention output f16, layout [B*T][1024], col = h*64+d
  float inv[8];
#pragma unroll
  for (int r = 0; r < 8; ++r) inv[r] = 1.0f / l_run[r];
#pragma unroll
  for (int nt = 0; nt < 4; ++nt)
#pragma unroll
    for (int r = 0; r < 8; ++r) {
      int t = bx * 64 + w * 16 + r + 8 * g;
      o[(size_t)(b * T + t) * Dm + h * Dh + nt * 16 + m16] =
          (_Float16)(oacc[nt][r] * inv[r]);
    }
}

extern "C" void kernel_launch(void* const* d_in, const int* in_sizes, int n_in,
                              void* d_out, int out_size, void* d_ws, size_t ws_size,
                              hipStream_t stream) {
  (void)in_sizes; (void)n_in; (void)out_size; (void)ws_size;
  const float* x    = (const float*)d_in[0];
  // d_in[1] = mask (causal, recomputed inline)
  const float* Wqkv = (const float*)d_in[2];
  const float* Wout = (const float*)d_in[3];
  float* out = (float*)d_out;

  constexpr int B = 4, T = 2048, D = 1024;
  constexpr int M = B * T;  // 8192
  _Float16* qkv  = (_Float16*)d_ws;                                                 // 48 MB
  _Float16* attn = (_Float16*)((char*)d_ws + (size_t)M * 3 * D * sizeof(_Float16)); // +16 MB

  // 1) qkv = x @ Wqkv   [8192 x 3072]
  gemm_wmma_kernel<float, _Float16>
      <<<dim3(3 * D / 128, M / 128), dim3(128), 0, stream>>>(x, Wqkv, qkv, M, 3 * D, D);

  // 2) causal flash attention per (64 q-rows, b*h)
  flash_attn_kernel<<<dim3(T / 64, B * 16), dim3(128), 0, stream>>>(qkv, attn);

  // 3) out = attn @ Wout   [8192 x 1024]
  gemm_wmma_kernel<_Float16, float>
      <<<dim3(D / 128, M / 128), dim3(128), 0, stream>>>(attn, Wout, out, M, D, D);
}
